// TransformerLayerShard_6579889897884
// MI455X (gfx1250) — matrix operand order repaired and checked by reading, not compiled, not run on hardware
//
#include <hip/hip_runtime.h>
#include <stdint.h>

// ---------------------------------------------------------------------------
// Problem constants (fixed by the reference)
// ---------------------------------------------------------------------------
#define BATCH 2
#define TSEQ  2048
#define DMODEL 2048
#define NHEAD 16
#define DHEAD 128
#define FFDIM 8192
#define MROWS (BATCH * TSEQ)          // 4096
#define QK_SCALE 0.08838834764831845f // 1/sqrt(128)

typedef __attribute__((ext_vector_type(16))) __bf16 v16bf;
typedef __attribute__((ext_vector_type(8)))  float  v8f;

struct Frag16 { union { v16bf v; uint4 q[2]; }; };

__device__ __forceinline__ float bf2f(uint16_t u) {
    union { uint32_t u; float f; } v; v.u = ((uint32_t)u) << 16; return v.f;
}
__device__ __forceinline__ uint16_t f2bf(float f) {
    union { float f; uint32_t u; } v; v.f = f;
    uint32_t u = v.u;
    u += 0x7fffu + ((u >> 16) & 1u);          // round-to-nearest-even
    return (uint16_t)(u >> 16);
}

// ---------------------------------------------------------------------------
// Weight transpose + fp32 -> bf16 convert:  dst[n*K + k] = bf16(src[k*N + n])
// ---------------------------------------------------------------------------
__global__ __launch_bounds__(256)
void transpose_cvt_kernel(const float* __restrict__ src, uint16_t* __restrict__ dst,
                          int Kd, int Nd) {
    __shared__ float tile[32][33];
    const int k0 = blockIdx.x * 32, n0 = blockIdx.y * 32;
    const int tx = threadIdx.x & 31, ty = threadIdx.x >> 5;
#pragma unroll
    for (int i = 0; i < 4; ++i) {
        int r = ty + i * 8;
        tile[r][tx] = src[(size_t)(k0 + r) * Nd + n0 + tx];
    }
    __syncthreads();
#pragma unroll
    for (int i = 0; i < 4; ++i) {
        int r = ty + i * 8;
        dst[(size_t)(n0 + r) * Kd + k0 + tx] = f2bf(tile[tx][r]);
    }
}

// ---------------------------------------------------------------------------
// LayerNorm: x bf16 [MROWS, D] -> xn bf16
// ---------------------------------------------------------------------------
__global__ __launch_bounds__(256)
void ln_kernel(const uint16_t* __restrict__ x, const float* __restrict__ g,
               const float* __restrict__ bta, uint16_t* __restrict__ xn) {
    __shared__ float rs[8], rss[8];
    const int row = blockIdx.x;
    const int tid = threadIdx.x;
    const uint16_t* xr = x + (size_t)row * DMODEL;
    float v[8], s = 0.f, ss = 0.f;
#pragma unroll
    for (int i = 0; i < 8; ++i) {
        v[i] = bf2f(xr[tid + i * 256]); s += v[i]; ss += v[i] * v[i];
    }
#pragma unroll
    for (int m = 16; m >= 1; m >>= 1) { s += __shfl_xor(s, m); ss += __shfl_xor(ss, m); }
    if ((tid & 31) == 0) { rs[tid >> 5] = s; rss[tid >> 5] = ss; }
    __syncthreads();
    float ts = 0.f, tss = 0.f;
#pragma unroll
    for (int i = 0; i < 8; ++i) { ts += rs[i]; tss += rss[i]; }
    const float mu  = ts * (1.0f / DMODEL);
    const float var = tss * (1.0f / DMODEL) - mu * mu;
    const float inv = rsqrtf(var + 1e-5f);
    uint16_t* xo = xn + (size_t)row * DMODEL;
#pragma unroll
    for (int i = 0; i < 8; ++i) {
        int d = tid + i * 256;
        xo[d] = f2bf((v[i] - mu) * inv * g[d] + bta[d]);
    }
}

// ---------------------------------------------------------------------------
// RoPE in place on q/k head-major [B,H,T,DH]; rotary first 64 dims, interleaved
// ---------------------------------------------------------------------------
__global__ __launch_bounds__(256)
void rope_kernel(uint16_t* __restrict__ qh, uint16_t* __restrict__ kh) {
    const int idx = blockIdx.x * 256 + threadIdx.x;   // B*H*T*32 threads
    const int p  = idx & 31;
    const int t  = (idx >> 5) & (TSEQ - 1);
    const int bh = idx >> 16;
    const int d0 = 2 * p, d1 = 2 * p + 1;
    const float pos = (float)t;
    // angle(d) = pos * 10000^(-2*(d mod 32)/64)
    const float kc = -0.28782313662425572f;           // -ln(10000)*2/64
    const float a0 = pos * __expf(kc * (float)(d0 & 31));
    const float a1 = pos * __expf(kc * (float)(d1 & 31));
    const size_t base = ((size_t)bh * TSEQ + t) * DHEAD;
    float c0 = cosf(a0), s0 = sinf(a0), c1 = cosf(a1), s1 = sinf(a1);
    {
        float x0 = bf2f(qh[base + d0]), x1 = bf2f(qh[base + d1]);
        qh[base + d0] = f2bf(x0 * c0 - x1 * s0);
        qh[base + d1] = f2bf(x1 * c1 + x0 * s1);
    }
    {
        float x0 = bf2f(kh[base + d0]), x1 = bf2f(kh[base + d1]);
        kh[base + d0] = f2bf(x0 * c0 - x1 * s0);
        kh[base + d1] = f2bf(x1 * c1 + x0 * s1);
    }
}

// ---------------------------------------------------------------------------
// WMMA GEMM: C[M,N] = A[M,K] (bf16, row-major) x BT[N,K]^T (bf16)
// Block tile 128x128, BK=32, double-buffered LDS (one barrier per K-step);
// 8 waves, each a 64x32 sub-tile (4x2 WMMA accumulators).
// Epilogue modes:
//   0: plain bf16 row-major store
//   1: head-major [B,H,T,DH] store, value *= scl   (Q with scl=1/sqrt(dh), K scl=1)
//   2: transposed head-major [B,H,DH,T] store      (V)
//   3: +bias, tanh-GELU, bf16 row-major            (W1)
//   4: +bias, + residual1 + residual2, bf16 -> out (W2 final)
// ---------------------------------------------------------------------------
__global__ __launch_bounds__(256)
void gemm_bf16_kernel(const uint16_t* __restrict__ A, const uint16_t* __restrict__ BT,
                      uint16_t* __restrict__ out, const float* __restrict__ bias,
                      const uint16_t* __restrict__ r1, const uint16_t* __restrict__ r2,
                      int N, int K, int mode, float scl) {
    __shared__ __align__(16) uint16_t As[2][128 * 32];
    __shared__ __align__(16) uint16_t Bs[2][128 * 32];
    const int tid = threadIdx.x;
    const int m0 = blockIdx.y * 128, n0 = blockIdx.x * 128;
    const int lane = tid & 31, wv = tid >> 5;
    const int wm = (wv >> 2) * 64, wn = (wv & 3) * 32;
    const int lm = lane & 15, lh = lane >> 4;

    // loader: 2 segments of 8 bf16 (16B) per thread per matrix
    const int sg0 = tid, sg1 = tid + 256;
    const int r0 = sg0 >> 2, c0 = (sg0 & 3) * 8, ofs0 = r0 * 32 + c0;
    const int r1i = sg1 >> 2, c1 = (sg1 & 3) * 8, ofs1 = r1i * 32 + c1;
    const uint16_t* Ap0 = A  + (size_t)(m0 + r0)  * K + c0;
    const uint16_t* Ap1 = A  + (size_t)(m0 + r1i) * K + c1;
    const uint16_t* Bp0 = BT + (size_t)(n0 + r0)  * K + c0;
    const uint16_t* Bp1 = BT + (size_t)(n0 + r1i) * K + c1;

    v8f acc[4][2];
#pragma unroll
    for (int mt = 0; mt < 4; ++mt)
#pragma unroll
        for (int nt = 0; nt < 2; ++nt)
#pragma unroll
            for (int e = 0; e < 8; ++e) acc[mt][nt][e] = 0.f;

    // prologue: stage K-tile 0 into buffer 0
    {
        uint4 ta0 = *(const uint4*)Ap0, ta1 = *(const uint4*)Ap1;
        uint4 tb0 = *(const uint4*)Bp0, tb1 = *(const uint4*)Bp1;
        *(uint4*)&As[0][ofs0] = ta0; *(uint4*)&As[0][ofs1] = ta1;
        *(uint4*)&Bs[0][ofs0] = tb0; *(uint4*)&Bs[0][ofs1] = tb1;
    }
    __syncthreads();

    int cur = 0;
    for (int kt = 32; kt <= K; kt += 32) {
        const bool more = (kt < K);
        uint4 ta0, ta1, tb0, tb1;
        if (more) {                 // global prefetch of next tile into registers
            ta0 = *(const uint4*)(Ap0 + kt); ta1 = *(const uint4*)(Ap1 + kt);
            tb0 = *(const uint4*)(Bp0 + kt); tb1 = *(const uint4*)(Bp1 + kt);
        }
        // compute current tile from LDS
        Frag16 af[4], bfr[2];
#pragma unroll
        for (int mt = 0; mt < 4; ++mt) {    // A frag: lane-half K{0-7,16-23}/{8-15,24-31}
            int rr = wm + mt * 16 + lm;
            af[mt].q[0] = *(const uint4*)&As[cur][rr * 32 + lh * 8];
            af[mt].q[1] = *(const uint4*)&As[cur][rr * 32 + lh * 8 + 16];
        }
#pragma unroll
        for (int nt = 0; nt < 2; ++nt) {    // B frag: lane-half K{0-7,16-23}/{8-15,24-31}
            int rr = wn + nt * 16 + lm;
            bfr[nt].q[0] = *(const uint4*)&Bs[cur][rr * 32 + lh * 8];
            bfr[nt].q[1] = *(const uint4*)&Bs[cur][rr * 32 + lh * 8 + 16];
        }
#pragma unroll
        for (int mt = 0; mt < 4; ++mt)
#pragma unroll
            for (int nt = 0; nt < 2; ++nt)
                acc[mt][nt] = __builtin_amdgcn_wmma_f32_16x16x32_bf16(
                    false, af[mt].v, false, bfr[nt].v, (short)0, acc[mt][nt], false, false);
        if (more) {                 // stage next tile into the other buffer
            *(uint4*)&As[cur ^ 1][ofs0] = ta0; *(uint4*)&As[cur ^ 1][ofs1] = ta1;
            *(uint4*)&Bs[cur ^ 1][ofs0] = tb0; *(uint4*)&Bs[cur ^ 1][ofs1] = tb1;
        }
        __syncthreads();            // single barrier per K-step
        cur ^= 1;
    }

#pragma unroll
    for (int mt = 0; mt < 4; ++mt)
#pragma unroll
        for (int nt = 0; nt < 2; ++nt) {
            const int col   = n0 + wn + nt * 16 + lm;
            const int rbase = m0 + wm + mt * 16 + lh * 8;
#pragma unroll
            for (int e = 0; e < 8; ++e) {
                const int row = rbase + e;
                float val = acc[mt][nt][e];
                if (mode == 0) {
                    out[(size_t)row * N + col] = f2bf(val);
                } else if (mode == 1) {
                    int bi = row >> 11, t = row & (TSEQ - 1);
                    int hh = col >> 7, dh = col & (DHEAD - 1);
                    out[(((size_t)(bi * NHEAD + hh) * TSEQ) + t) * DHEAD + dh] = f2bf(val * scl);
                } else if (mode == 2) {
                    int bi = row >> 11, t = row & (TSEQ - 1);
                    int hh = col >> 7, dh = col & (DHEAD - 1);
                    out[(((size_t)(bi * NHEAD + hh) * DHEAD) + dh) * TSEQ + t] = f2bf(val);
                } else if (mode == 3) {
                    val += bias[col];
                    float u = 0.7978845608028654f * (val + 0.044715f * val * val * val);
                    val = 0.5f * val * (1.0f + tanhf(u));
                    out[(size_t)row * N + col] = f2bf(val);
                } else {
                    val += bias[col];
                    size_t ix = (size_t)row * N + col;
                    out[ix] = f2bf(val + bf2f(r1[ix]) + bf2f(r2[ix]));
                }
            }
        }
}

// ---------------------------------------------------------------------------
// Flash attention: per (b,h, 128-q tile). Q frags in registers, K/V^T in LDS,
// online softmax, P through wave-private LDS, O += P@V via WMMA.
// Next K/V tile prefetched (global_prefetch_b8) during compute.
// q/k: [B,H,T,DH] bf16 (q pre-scaled). vt: [B,H,DH,T] bf16. ctx: [B,T,D] bf16.
// ---------------------------------------------------------------------------
__global__ __launch_bounds__(256)
void flash_attn_kernel(const uint16_t* __restrict__ qh, const uint16_t* __restrict__ kh,
                       const uint16_t* __restrict__ vt, uint16_t* __restrict__ ctx) {
    __shared__ __align__(16) uint16_t Ks[64 * 128];   // [kv][dh]
    __shared__ __align__(16) uint16_t Vs[128 * 64];   // [dh][kv]
    __shared__ __align__(16) uint16_t Ps[8 * 16 * 64];// per-wave P staging
    const int tid = threadIdx.x, lane = tid & 31, wv = tid >> 5;
    const int lm = lane & 15, lh = lane >> 4;
    const int b = blockIdx.z, h = blockIdx.y, q0 = blockIdx.x * 128;
    const size_t headTD = (size_t)(b * NHEAD + h) * TSEQ * DHEAD;

    Frag16 qf[4];
    {
        const uint16_t* qp = qh + headTD + (size_t)(q0 + wv * 16 + lm) * DHEAD;
#pragma unroll
        for (int c = 0; c < 4; ++c) {         // A-layout fragments
            qf[c].q[0] = *(const uint4*)(qp + c * 32 + lh * 8);
            qf[c].q[1] = *(const uint4*)(qp + c * 32 + lh * 8 + 16);
        }
    }
    v8f o8[8];
#pragma unroll
    for (int nt = 0; nt < 8; ++nt)
#pragma unroll
        for (int e = 0; e < 8; ++e) o8[nt][e] = 0.f;
    float mrun[8], lrun[8];
#pragma unroll
    for (int e = 0; e < 8; ++e) { mrun[e] = -1e30f; lrun[e] = 0.f; }

    for (int kv0 = 0; kv0 < TSEQ; kv0 += 64) {
#pragma unroll
        for (int i = 0; i < 4; ++i) {
            int seg = tid + i * 256;          // 1024 segments of 16B each
            int rk = seg >> 4, sk = (seg & 15) * 8;   // K tile: 64 rows x 16 segs
            *(uint4*)&Ks[rk * 128 + sk] =
                *(const uint4*)(kh + headTD + (size_t)(kv0 + rk) * DHEAD + sk);
            int rv = seg >> 3, sv = (seg & 7) * 8;    // V^T tile: 128 rows x 8 segs
            *(uint4*)&Vs[rv * 64 + sv] =
                *(const uint4*)(vt + headTD + (size_t)rv * TSEQ + kv0 + sv);
        }
        __syncthreads();

        // prefetch next K/V tile into cache while we compute (global_prefetch_b8)
        if (kv0 + 64 < TSEQ) {
            const int nkv = kv0 + 64;
            __builtin_prefetch(kh + headTD + (size_t)(nkv + (tid >> 2)) * DHEAD +
                               (tid & 3) * 32, 0, 1);
            __builtin_prefetch(vt + headTD + (size_t)(tid >> 1) * TSEQ + nkv +
                               (tid & 1) * 32, 0, 1);
        }

        // S = Q K^T  (16q x 64kv per wave)
        v8f s4[4];
#pragma unroll
        for (int kvn = 0; kvn < 4; ++kvn)
#pragma unroll
            for (int e = 0; e < 8; ++e) s4[kvn][e] = 0.f;
#pragma unroll
        for (int c = 0; c < 4; ++c)
#pragma unroll
            for (int kvn = 0; kvn < 4; ++kvn) {
                Frag16 kb;
                int krow = kvn * 16 + lm;     // B frag: table-order K along dh
                kb.q[0] = *(const uint4*)&Ks[krow * 128 + c * 32 + lh * 8];
                kb.q[1] = *(const uint4*)&Ks[krow * 128 + c * 32 + lh * 8 + 16];
                s4[kvn] = __builtin_amdgcn_wmma_f32_16x16x32_bf16(
                    false, qf[c].v, false, kb.v, (short)0, s4[kvn], false, false);
            }

        // online softmax: rows live in a 16-lane half -> shfl-xor masks 1,2,4,8
        float tmax[8], sc[8], rsum[8];
#pragma unroll
        for (int e = 0; e < 8; ++e)
            tmax[e] = fmaxf(fmaxf(s4[0][e], s4[1][e]), fmaxf(s4[2][e], s4[3][e]));
#pragma unroll
        for (int e = 0; e < 8; ++e)
#pragma unroll
            for (int m2 = 1; m2 < 16; m2 <<= 1)
                tmax[e] = fmaxf(tmax[e], __shfl_xor(tmax[e], m2));
#pragma unroll
        for (int e = 0; e < 8; ++e) {
            float nm = fmaxf(mrun[e], tmax[e]);
            sc[e] = __expf(mrun[e] - nm);
            mrun[e] = nm; rsum[e] = 0.f;
        }
#pragma unroll
        for (int kvn = 0; kvn < 4; ++kvn)
#pragma unroll
            for (int e = 0; e < 8; ++e) {
                float p = __expf(s4[kvn][e] - mrun[e]);
                s4[kvn][e] = p; rsum[e] += p;
            }
#pragma unroll
        for (int e = 0; e < 8; ++e)
#pragma unroll
            for (int m2 = 1; m2 < 16; m2 <<= 1)
                rsum[e] += __shfl_xor(rsum[e], m2);
#pragma unroll
        for (int e = 0; e < 8; ++e) lrun[e] = lrun[e] * sc[e] + rsum[e];
#pragma unroll
        for (int nt = 0; nt < 8; ++nt)
#pragma unroll
            for (int e = 0; e < 8; ++e) o8[nt][e] *= sc[e];

        // stage P (C-layout -> A-layout via wave-private LDS; in-order DS ops)
        const int pb = wv * 1024;
#pragma unroll
        for (int kvn = 0; kvn < 4; ++kvn)
#pragma unroll
            for (int e = 0; e < 8; ++e)
                Ps[pb + (e + lh * 8) * 64 + kvn * 16 + lm] = f2bf(s4[kvn][e]);

        // O += P @ V
#pragma unroll
        for (int kc = 0; kc < 2; ++kc) {
            Frag16 pa;
            pa.q[0] = *(const uint4*)&Ps[pb + lm * 64 + kc * 32 + lh * 8];
            pa.q[1] = *(const uint4*)&Ps[pb + lm * 64 + kc * 32 + lh * 8 + 16];
#pragma unroll
            for (int nt = 0; nt < 8; ++nt) {
                Frag16 vb;
                int vrow = nt * 16 + lm;
                vb.q[0] = *(const uint4*)&Vs[vrow * 64 + kc * 32 + lh * 8];
                vb.q[1] = *(const uint4*)&Vs[vrow * 64 + kc * 32 + lh * 8 + 16];
                o8[nt] = __builtin_amdgcn_wmma_f32_16x16x32_bf16(
                    false, pa.v, false, vb.v, (short)0, o8[nt], false, false);
            }
        }
        __syncthreads();
    }

#pragma unroll
    for (int nt = 0; nt < 8; ++nt)
#pragma unroll
        for (int e = 0; e < 8; ++e) {
            int t  = q0 + wv * 16 + e + lh * 8;
            int dh = nt * 16 + lm;
            float val = o8[nt][e] / lrun[e];
            ctx[((size_t)(b * TSEQ + t)) * DMODEL + h * DHEAD + dh] = f2bf(val);
        }
}

// ---------------------------------------------------------------------------
// Launch sequence
// ---------------------------------------------------------------------------
extern "C" void kernel_launch(void* const* d_in, const int* in_sizes, int n_in,
                              void* d_out, int out_size, void* d_ws, size_t ws_size,
                              hipStream_t stream) {
    (void)in_sizes; (void)n_in; (void)out_size; (void)ws_size;
    const uint16_t* x  = (const uint16_t*)d_in[0];
    const float* Wq = (const float*)d_in[1];
    const float* Wk = (const float*)d_in[2];
    const float* Wv = (const float*)d_in[3];
    const float* Wo = (const float*)d_in[4];
    const float* W1 = (const float*)d_in[5];
    const float* b1 = (const float*)d_in[6];
    const float* W2 = (const float*)d_in[7];
    const float* b2 = (const float*)d_in[8];
    const float* lns = (const float*)d_in[9];
    const float* lnb = (const float*)d_in[10];
    uint16_t* out = (uint16_t*)d_out;

    char* ws = (char*)d_ws;
    size_t off = 0;
    auto alloc = [&](size_t bytes) -> void* {
        void* p = ws + off;
        off += (bytes + 255) & ~(size_t)255;
        return p;
    };
    const size_t SZ_DD  = (size_t)DMODEL * DMODEL * 2;  // 8 MB
    const size_t SZ_DF  = (size_t)DMODEL * FFDIM  * 2;  // 33.5 MB
    const size_t SZ_ACT = (size_t)MROWS  * DMODEL * 2;  // 16.8 MB
    uint16_t* WqT = (uint16_t*)alloc(SZ_DD);
    uint16_t* WkT = (uint16_t*)alloc(SZ_DD);
    uint16_t* WvT = (uint16_t*)alloc(SZ_DD);
    uint16_t* WoT = (uint16_t*)alloc(SZ_DD);
    uint16_t* W1T = (uint16_t*)alloc(SZ_DF);
    uint16_t* W2T = (uint16_t*)alloc(SZ_DF);
    uint16_t* xn   = (uint16_t*)alloc(SZ_ACT);
    uint16_t* attn = (uint16_t*)alloc(SZ_ACT);
    // q/k/vT/ctx block; 67 MB GELU buffer aliases it after attention is done
    uint16_t* qhm = (uint16_t*)alloc(SZ_ACT);
    uint16_t* khm = (uint16_t*)alloc(SZ_ACT);
    uint16_t* vtm = (uint16_t*)alloc(SZ_ACT);
    uint16_t* ctx = (uint16_t*)alloc(SZ_ACT);
    uint16_t* hg  = qhm;                         // [MROWS, FFDIM] bf16 = 4*SZ_ACT

    // 1) weight convert+transpose
    transpose_cvt_kernel<<<dim3(DMODEL/32, DMODEL/32), 256, 0, stream>>>(Wq, WqT, DMODEL, DMODEL);
    transpose_cvt_kernel<<<dim3(DMODEL/32, DMODEL/32), 256, 0, stream>>>(Wk, WkT, DMODEL, DMODEL);
    transpose_cvt_kernel<<<dim3(DMODEL/32, DMODEL/32), 256, 0, stream>>>(Wv, WvT, DMODEL, DMODEL);
    transpose_cvt_kernel<<<dim3(DMODEL/32, DMODEL/32), 256, 0, stream>>>(Wo, WoT, DMODEL, DMODEL);
    transpose_cvt_kernel<<<dim3(DMODEL/32, FFDIM/32),  256, 0, stream>>>(W1, W1T, DMODEL, FFDIM);
    transpose_cvt_kernel<<<dim3(FFDIM/32,  DMODEL/32), 256, 0, stream>>>(W2, W2T, FFDIM, DMODEL);
    // 2) layernorm
    ln_kernel<<<MROWS, 256, 0, stream>>>(x, lns, lnb, xn);
    // 3) QKV projections (Q scaled; V stored transposed per head)
    gemm_bf16_kernel<<<dim3(DMODEL/128, MROWS/128), 256, 0, stream>>>(
        xn, WqT, qhm, nullptr, nullptr, nullptr, DMODEL, DMODEL, 1, QK_SCALE);
    gemm_bf16_kernel<<<dim3(DMODEL/128, MROWS/128), 256, 0, stream>>>(
        xn, WkT, khm, nullptr, nullptr, nullptr, DMODEL, DMODEL, 1, 1.0f);
    gemm_bf16_kernel<<<dim3(DMODEL/128, MROWS/128), 256, 0, stream>>>(
        xn, WvT, vtm, nullptr, nullptr, nullptr, DMODEL, DMODEL, 2, 1.0f);
    // 4) RoPE
    rope_kernel<<<(BATCH * NHEAD * TSEQ * 32) / 256, 256, 0, stream>>>(qhm, khm);
    // 5) flash attention
    flash_attn_kernel<<<dim3(TSEQ/128, NHEAD, BATCH), 256, 0, stream>>>(qhm, khm, vtm, ctx);
    // 6) output projection
    gemm_bf16_kernel<<<dim3(DMODEL/128, MROWS/128), 256, 0, stream>>>(
        ctx, WoT, attn, nullptr, nullptr, nullptr, DMODEL, DMODEL, 0, 1.0f);
    // 7) MLP up + GELU (overwrites dead q/k/vT/ctx region)
    gemm_bf16_kernel<<<dim3(FFDIM/128, MROWS/128), 256, 0, stream>>>(
        xn, W1T, hg, b1, nullptr, nullptr, FFDIM, DMODEL, 3, 1.0f);
    // 8) MLP down + bias + residual(x) + residual(attn) -> final output
    gemm_bf16_kernel<<<dim3(DMODEL/128, MROWS/128), 256, 0, stream>>>(
        hg, W2T, out, b2, x, attn, DMODEL, FFDIM, 4, 1.0f);
}
